// QNet_11450382811781
// MI455X (gfx1250) — compile-verified, matrix operand order checked by reading, not run on hardware
//
#include <hip/hip_runtime.h>
#include <hip/hip_bf16.h>

// B=16384, A=32, NOBS=2, H1=128, HX=64, NACT=5
// scales = [GRID_X-1, GRID_Y-1] = [2, 6]; NEIGHBORHOOD = 2

typedef __attribute__((ext_vector_type(16))) _Float16 v16h;
typedef __attribute__((ext_vector_type(8)))  _Float16 v8h;
typedef __attribute__((ext_vector_type(8)))  float    v8f;
typedef __attribute__((ext_vector_type(2)))  float    v2f;

#define DI __device__ __forceinline__

#if __has_builtin(__builtin_amdgcn_wmma_f32_16x16x4_f32)
#define HAVE_WMMA_F32X4 1
#else
#define HAVE_WMMA_F32X4 0
#endif

DI v16h make_frag(const _Float16* lo, const _Float16* hi) {
    union { v16h v; v8h h[2]; } u;
    u.h[0] = *(const v8h*)lo;
    u.h[1] = *(const v8h*)hi;
    return u.v;
}

DI v8f wmma_f16(v16h a, v16h b, v8f c) {
    // D = A(16x32 f16) * B(32x16 f16) + C(16x16 f32)
    return __builtin_amdgcn_wmma_f32_16x16x32_f16(
        false, a, false, b, (short)0, c, false, false);
}

#if HAVE_WMMA_F32X4
DI v8f wmma_f32x4(v2f a, v2f b, v8f c) {
    // D = A(16x4 f32) * B(4x16 f32) + C(16x16 f32)
    return __builtin_amdgcn_wmma_f32_16x16x4_f32(
        false, a, false, b, (short)0, c, false, false);
}
#endif

// ---------------------------------------------------------------------------
// Pack weights into WMMA B-operand fragments.
//   W2p: f16 [32 agents][4 ntile][4 kchunk][32 lane][16 elem]   (512 frags)
//   Wcp: f16 same shape                                          (512 frags)
//   Wdp: f16 [32 agents][2 kchunk][32 lane][16 elem], cols>=5 zero (64 frags)
//   W1p: f32 [32 agents][8 ntile][32 lane][2 elem], lanes>=16 zero (256 frags)
// f16 B layout mirrors the 16-bit A layout with N in the "row" role:
//   col = ntile*16 + (lane&15); kb=(lane>>4)*8;
//   elem e -> K = kchunk*32 + (e<8 ? kb+e : 16+kb+(e-8))
// f32 B (4x16): lane holds col; lanes 0-15 -> K=0,1; lanes 16-31 -> K=2,3 (=0)
// ---------------------------------------------------------------------------
__global__ void pack_weights(const float* __restrict__ W1,
                             const float* __restrict__ W2,
                             const float* __restrict__ Wc,
                             const float* __restrict__ Wd,
                             _Float16* __restrict__ W2p,
                             _Float16* __restrict__ Wcp,
                             _Float16* __restrict__ Wdp,
                             float*    __restrict__ W1p) {
    int t = blockIdx.x * blockDim.x + threadIdx.x;
    const int TOT = (512 + 512 + 64 + 256) * 32;
    if (t >= TOT) return;
    int frag = t >> 5, lane = t & 31;
    int cl = lane & 15, kb = (lane >> 4) * 8;
    if (frag < 1024) {
        const float* W = (frag < 512) ? W2 : Wc;
        _Float16*    P = (frag < 512) ? W2p : Wcp;
        int fid = frag & 511;
        int a = fid >> 4, n = (fid >> 2) & 3, c = fid & 3;
        int col = n * 16 + cl;
        _Float16* dst = P + (size_t)fid * 512 + lane * 16;
#pragma unroll
        for (int e = 0; e < 16; ++e) {
            int k = c * 32 + (e < 8 ? kb + e : 16 + kb + (e - 8));
            dst[e] = (_Float16)W[(size_t)a * 8192 + k * 64 + col]; // [A][128][64]
        }
    } else if (frag < 1088) {
        int fid = frag - 1024;
        int a = fid >> 1, c = fid & 1;
        _Float16* dst = Wdp + (size_t)fid * 512 + lane * 16;
#pragma unroll
        for (int e = 0; e < 16; ++e) {
            int k = c * 32 + (e < 8 ? kb + e : 16 + kb + (e - 8)); // k < 64
            float v = (cl < 5) ? Wd[(size_t)a * 320 + k * 5 + cl] : 0.0f; // [A][64][5]
            dst[e] = (_Float16)v;
        }
    } else {
        int fid = frag - 1088;                 // 0..255
        int a = fid >> 3, nt = fid & 7;
        int col = nt * 16 + cl;
        float* dst = W1p + ((size_t)fid * 32 + lane) * 2;
        dst[0] = (lane < 16) ? W1[a * 256 + col]       : 0.0f; // K=0 / K=2(pad)
        dst[1] = (lane < 16) ? W1[a * 256 + 128 + col] : 0.0f; // K=1 / K=3(pad)
    }
}

// ---------------------------------------------------------------------------
// Main fused kernel: one block = 16 batch rows x all 32 agents, 4 waves.
// ---------------------------------------------------------------------------
__global__ __launch_bounds__(128) void qnet_main(
    const float* __restrict__ obs,   // [B][32][2]
    const float* __restrict__ W1,    // [32][2][128] (fallback path only)
    const float* __restrict__ b1,    // [32][128]
    const float* __restrict__ b2,    // [32][64]
    const float* __restrict__ bc,    // [32][64]
    const float* __restrict__ bd,    // [32][5]
    const _Float16* __restrict__ W2p,
    const _Float16* __restrict__ Wcp,
    const _Float16* __restrict__ Wdp,
    const float* __restrict__ W1p,
    float* __restrict__ out)         // [B][32][5]
{
    extern __shared__ __align__(16) char smem[];
    float*    ox   = (float*)smem;             // [16][32]
    float*    oy   = ox + 512;                 // [16][32]
    float*    jinv = oy + 512;                 // [16][32]
    _Float16* h1   = (_Float16*)(jinv + 512);  // [2][16][128] double-buffered
    _Float16* hid  = h1 + 2 * 16 * 128;        // [16][32][64]  (b, agent, h)
    _Float16* hidT = hid + 16 * 32 * 64;       // [16][64][32]  (b, h, agent)
    _Float16* cms  = hidT + 16 * 64 * 32;      // [16][32][64]
    _Float16* h2   = cms + 16 * 32 * 64;       // [4][16][64]

    const int t    = threadIdx.x;
    const int lane = t & 31;
    const int w    = t >> 5;           // wave id 0..3
    const int cl   = lane & 15;        // A-frag row / C col / B-frag col
    const int kb   = (lane >> 4) * 8;  // K sub-offset per ISA layout
    const int rhi  = (lane >> 4) * 8;  // C/D row base
    const int b0   = blockIdx.x * 16;

    // ---- load obs tile -> LDS (split x/y) ----
#pragma unroll
    for (int k = 0; k < 8; ++k) {
        int idx = t + k * 128;                 // 0..1023
        int b = idx >> 6, r = idx & 63, a = r >> 1, comp = r & 1;
        float v = obs[(size_t)(b0 + b) * 64 + a * 2 + comp];
        if (comp == 0) ox[b * 32 + a] = v; else oy[b * 32 + a] = v;
    }
    __syncthreads();

    // ---- 1/J per (b, i) ----
    for (int p = t; p < 512; p += 128) {
        int b = p >> 5, i = p & 31;
        float xi = ox[b * 32 + i], yi = oy[b * 32 + i];
        float s = 0.f;
#pragma unroll
        for (int j = 0; j < 32; ++j) {
            float d = 2.0f * fabsf(xi - ox[b * 32 + j]) +
                      6.0f * fabsf(yi - oy[b * 32 + j]);
            s += (d > 0.0f && d < 2.0f) ? 1.0f : 0.0f;
        }
        jinv[p] = 1.0f / fmaxf(s, 1.0f);
    }

    // ---- encoder: agents processed in pairs for independent WMMA chains ----
    for (int a0 = 0; a0 < 32; a0 += 2) {
        __syncthreads();
        // stage 1: h1[ag] = relu(obs @ W1[a0+ag] + b1)  (K=2 padded to 4)
#if HAVE_WMMA_F32X4
        {
            v2f af;
            af.x = (lane < 16) ? ox[cl * 32 + 0] : 0.0f; // patched per-agent below
            af.y = (lane < 16) ? oy[cl * 32 + 0] : 0.0f;
#pragma unroll
            for (int ag = 0; ag < 2; ++ag) {
                int a = a0 + ag;
                v2f aa;
                aa.x = (lane < 16) ? ox[cl * 32 + a] : 0.0f;
                aa.y = (lane < 16) ? oy[cl * 32 + a] : 0.0f;
#pragma unroll
                for (int q = 0; q < 2; ++q) {
                    int nt = w * 2 + q;                     // 8 ntiles over 4 waves
                    const v2f bf = *(const v2f*)(W1p + ((size_t)(a * 8 + nt) * 32 + lane) * 2);
                    v8f acc = {};
                    acc = wmma_f32x4(aa, bf, acc);
                    int n = nt * 16 + cl;
                    float bias = b1[a * 128 + n];
#pragma unroll
                    for (int r = 0; r < 8; ++r) {
                        int br = rhi + r;
                        h1[ag * 2048 + br * 128 + n] =
                            (_Float16)fmaxf(acc[r] + bias, 0.0f);
                    }
                }
            }
            (void)af;
        }
#else
        {   // fallback: thread t owns column o=t
#pragma unroll
            for (int ag = 0; ag < 2; ++ag) {
                int a = a0 + ag;
                float w10 = W1[a * 256 + t];
                float w11 = W1[a * 256 + 128 + t];
                float bb  = b1[a * 128 + t];
#pragma unroll
                for (int b = 0; b < 16; ++b) {
                    float v = ox[b * 32 + a] * w10 + oy[b * 32 + a] * w11 + bb;
                    h1[ag * 2048 + b * 128 + t] = (_Float16)fmaxf(v, 0.0f);
                }
            }
        }
#endif
        __syncthreads();
        // stage 2: hidden = relu(h1 @ W2 + b2); two agents interleaved
        {
            v8f acc0 = {}, acc1 = {};
#pragma unroll
            for (int c = 0; c < 4; ++c) {
                v16h af0 = make_frag(&h1[cl * 128 + c * 32 + kb],
                                     &h1[cl * 128 + c * 32 + 16 + kb]);
                v16h af1 = make_frag(&h1[2048 + cl * 128 + c * 32 + kb],
                                     &h1[2048 + cl * 128 + c * 32 + 16 + kb]);
                const _Float16* bp0 = W2p + ((size_t)((a0 * 4 + w) * 4 + c) * 32 + lane) * 16;
                const _Float16* bp1 = W2p + ((size_t)(((a0 + 1) * 4 + w) * 4 + c) * 32 + lane) * 16;
                v16h bf0 = make_frag(bp0, bp0 + 8);
                v16h bf1 = make_frag(bp1, bp1 + 8);
                acc0 = wmma_f16(af0, bf0, acc0);
                acc1 = wmma_f16(af1, bf1, acc1);
            }
            int n = w * 16 + cl;
#pragma unroll
            for (int ag = 0; ag < 2; ++ag) {
                int a = a0 + ag;
                float bias = b2[a * 64 + n];
                const v8f& acc = ag ? acc1 : acc0;
#pragma unroll
                for (int r = 0; r < 8; ++r) {
                    int br = rhi + r;
                    _Float16 hv = (_Float16)fmaxf(acc[r] + bias, 0.0f);
                    hid [(br * 32 + a) * 64 + n] = hv;
                    hidT[(br * 64 + n) * 32 + a] = hv;
                }
            }
        }
    }
    __syncthreads();

    // ---- neighbor mixing: per b, total = mask(32x32) @ hidden(32x64) ----
    for (int bi = 0; bi < 4; ++bi) {
        int b = w * 4 + bi;                    // wave owns 4 batch rows
#pragma unroll
        for (int it = 0; it < 2; ++it) {       // agent-row tiles
            int i = it * 16 + cl;
            float xi = ox[b * 32 + i], yi = oy[b * 32 + i];
            v16h af;                           // mask fragment, built in-register
#pragma unroll
            for (int e = 0; e < 16; ++e) {
                int j = (e < 8) ? (kb + e) : (16 + kb + (e - 8));
                float d = 2.0f * fabsf(xi - ox[b * 32 + j]) +
                          6.0f * fabsf(yi - oy[b * 32 + j]);
                af[e] = (d > 0.0f && d < 2.0f) ? (_Float16)1.0f : (_Float16)0.0f;
            }
#pragma unroll
            for (int nt = 0; nt < 4; ++nt) {
                int h = nt * 16 + cl;
                const _Float16* hp = &hidT[(b * 64 + h) * 32];
                v16h bf = make_frag(hp + kb, hp + 16 + kb);
                v8f acc = {};
                acc = wmma_f16(af, bf, acc);
#pragma unroll
                for (int r = 0; r < 8; ++r) {
                    int ir = it * 16 + rhi + r;
                    float v = acc[r] * jinv[b * 32 + ir];
                    cms[(b * 32 + ir) * 64 + h] = (_Float16)v;
                }
            }
        }
    }
    __syncthreads();

    // ---- comm layer (tanh) + decoder, 4 agents per group ----
    for (int g = 0; g < 8; ++g) {
#pragma unroll
        for (int ap = 0; ap < 2; ++ap) {       // two agent-pairs, interleaved chains
            int a0 = g * 4 + ap * 2;
            v8f acc0 = {}, acc1 = {};
#pragma unroll
            for (int c = 0; c < 4; ++c) {      // K 0..63: hidden, 64..127: comms
                const _Float16* src = (c < 2) ? hid : cms;
                const _Float16* ap0 = src + ((size_t)(cl * 32 + a0) * 64) + (c & 1) * 32;
                const _Float16* ap1 = src + ((size_t)(cl * 32 + a0 + 1) * 64) + (c & 1) * 32;
                v16h af0 = make_frag(ap0 + kb, ap0 + 16 + kb);
                v16h af1 = make_frag(ap1 + kb, ap1 + 16 + kb);
                const _Float16* bp0 = Wcp + ((size_t)((a0 * 4 + w) * 4 + c) * 32 + lane) * 16;
                const _Float16* bp1 = Wcp + ((size_t)(((a0 + 1) * 4 + w) * 4 + c) * 32 + lane) * 16;
                v16h bf0 = make_frag(bp0, bp0 + 8);
                v16h bf1 = make_frag(bp1, bp1 + 8);
                acc0 = wmma_f16(af0, bf0, acc0);
                acc1 = wmma_f16(af1, bf1, acc1);
            }
            int n = w * 16 + cl;
#pragma unroll
            for (int ag = 0; ag < 2; ++ag) {
                int a = a0 + ag;
                int aa = ap * 2 + ag;
                float bias = bc[a * 64 + n];
                const v8f& acc = ag ? acc1 : acc0;
#pragma unroll
                for (int r = 0; r < 8; ++r) {
                    float v = tanhf(acc[r] + bias);
                    h2[(aa * 16 + rhi + r) * 64 + n] = (_Float16)v;
                }
            }
        }
        __syncthreads();
        {   // decoder: wave w -> agent g*4+w, 16x16x64 (N padded 5->16)
            int a = g * 4 + w;
            v8f acc = {};
#pragma unroll
            for (int c = 0; c < 2; ++c) {
                const _Float16* ap = &h2[(w * 16 + cl) * 64 + c * 32];
                v16h af = make_frag(ap + kb, ap + 16 + kb);
                const _Float16* bp = Wdp + ((size_t)((a * 2 + c) * 32 + lane) * 16);
                v16h bf = make_frag(bp, bp + 8);
                acc = wmma_f16(af, bf, acc);
            }
            if (cl < 5) {
                float bias = bd[a * 5 + cl];
#pragma unroll
                for (int r = 0; r < 8; ++r) {
                    int br = rhi + r;
                    out[((size_t)(b0 + br) * 32 + a) * 5 + cl] = acc[r] + bias;
                }
            }
        }
        __syncthreads();
    }
}

extern "C" void kernel_launch(void* const* d_in, const int* in_sizes, int n_in,
                              void* d_out, int out_size, void* d_ws, size_t ws_size,
                              hipStream_t stream) {
    (void)in_sizes; (void)n_in; (void)out_size; (void)ws_size;
    const float* obs = (const float*)d_in[0];
    const float* W1  = (const float*)d_in[1];
    const float* b1  = (const float*)d_in[2];
    const float* W2  = (const float*)d_in[3];
    const float* b2  = (const float*)d_in[4];
    const float* Wc  = (const float*)d_in[5];
    const float* bc  = (const float*)d_in[6];
    const float* Wd  = (const float*)d_in[7];
    const float* bd  = (const float*)d_in[8];
    float* out = (float*)d_out;

    // workspace carve
    _Float16* W2p = (_Float16*)d_ws;
    _Float16* Wcp = W2p + (size_t)512 * 512;
    _Float16* Wdp = Wcp + (size_t)512 * 512;
    float*    W1p = (float*)(Wdp + (size_t)64 * 512);   // 256 frags * 64 floats

    int packThreads = (512 + 512 + 64 + 256) * 32;      // 43008
    pack_weights<<<(packThreads + 127) / 128, 128, 0, stream>>>(
        W1, W2, Wc, Wd, W2p, Wcp, Wdp, W1p);

    // dynamic LDS: 3*2KB f32 + 8KB h1 + 3*64KB f16 + 8KB h2 = 219136 B (<320KB/WGP)
    size_t lds = 512 * 4 * 3 + 2 * 16 * 128 * 2 + 3 * (16 * 32 * 64 * 2) + 4 * 16 * 64 * 2;
    qnet_main<<<16384 / 16, 128, lds, stream>>>(obs, W1, b1, b2, bc, bd,
                                                W2p, Wcp, Wdp, W1p, out);
}
